// SimpleFineGrainedHead2_63883343561448
// MI455X (gfx1250) — matrix operand order repaired and checked by reading, not compiled, not run on hardware
//
#include <hip/hip_runtime.h>
#include <math.h>

// ---------------- problem constants ----------------
#define BN 384
#define QN 384
#define VN 197
#define TN 100
#define DD 768
#define EE 256
#define KK 50
#define KPAD 64

// cross-sim tiling
#define GB 4          // b-tiles per block
#define GQ 4          // q-tiles per block
#define KC 128        // K-chunk staged in LDS
#define CS_THREADS 512

typedef __attribute__((ext_vector_type(16))) __bf16 bf16x16;
typedef __attribute__((ext_vector_type(8)))  float  f32x8;

// ---------------- WMMA helpers ----------------
__device__ inline f32x8 wmma_bf16(bf16x16 a, bf16x16 b, f32x8 c) {
  // D = A(16x32 bf16) * B(32x16 bf16) + C(16x16 f32)
  return __builtin_amdgcn_wmma_f32_16x16x32_bf16(
      /*neg_a=*/false, a, /*neg_b=*/false, b,
      /*c_mod=*/(short)0, c, /*reuse_a=*/false, /*reuse_b=*/false);
}

// A fragment (16x32, MxK) from row-major f32 source, convert to bf16 in-register.
// Lane L: row = L&15 ; kbase = (L>>4)*8 ; v[0..3] = K kbase..kbase+7 ; v[4..7] = K kbase+16..kbase+23
__device__ inline bf16x16 load_a_f32(const float* __restrict__ src, int ld, int row0, int k0) {
  int lane = threadIdx.x & 31;
  const float* p = src + (size_t)(row0 + (lane & 15)) * ld + k0 + ((lane >> 4) << 3);
  bf16x16 a;
#pragma unroll
  for (int i = 0; i < 8; ++i) a[i] = (__bf16)p[i];
#pragma unroll
  for (int i = 0; i < 8; ++i) a[8 + i] = (__bf16)p[16 + i];
  return a;
}

// A fragment from row-major bf16 source (global or LDS; fully inlined).
__device__ inline bf16x16 load_a_bf16(const __bf16* src, int ld, int row0, int k0) {
  int lane = threadIdx.x & 31;
  const __bf16* p = src + (size_t)(row0 + (lane & 15)) * ld + k0 + ((lane >> 4) << 3);
  bf16x16 a;
#pragma unroll
  for (int i = 0; i < 8; ++i) a[i] = p[i];
#pragma unroll
  for (int i = 0; i < 8; ++i) a[8 + i] = p[16 + i];
  return a;
}

// B fragment (32x16, KxN) where the source is "N-major": Bsrc[n][k] row-major with stride ld.
// Lane L: n = L&15 ; kbase = (L>>4)*8 ; b[0..7] = K kbase..kbase+7 ; b[8..15] = K kbase+16..kbase+23
__device__ inline bf16x16 load_b_bf16(const __bf16* src, int ld, int n0, int k0) {
  int lane = threadIdx.x & 31;
  const __bf16* p = src + (size_t)(n0 + (lane & 15)) * ld + k0 + ((lane >> 4) << 3);
  bf16x16 b;
#pragma unroll
  for (int i = 0; i < 8; ++i) b[i] = p[i];
#pragma unroll
  for (int i = 0; i < 8; ++i) b[8 + i] = p[16 + i];
  return b;
}

__device__ inline float wave_sum(float v) {
#pragma unroll
  for (int off = 16; off > 0; off >>= 1) v += __shfl_xor(v, off, 32);
  return v;
}

// ---------------- kernels ----------------

// Transposed bf16 weight: Wt[n*768 + k] = bf16(W[k*256 + n])
__global__ void k_transpose_w(const float* __restrict__ W, __bf16* __restrict__ Wt) {
  int idx = blockIdx.x * blockDim.x + threadIdx.x;
  if (idx >= EE * DD) return;
  int n = idx / DD, k = idx % DD;
  Wt[idx] = (__bf16)W[(size_t)k * EE + n];
}

// Y[M,256] = X[M,768] @ W + bias ; W provided transposed-bf16 [256,768].
// block=128 (4 waves); each block = one 16-row M tile; wave w covers N in [w*64, w*64+64).
__global__ void k_gemm_proj(const float* __restrict__ X, const __bf16* __restrict__ Wt,
                            const float* __restrict__ bias, float* __restrict__ Y, int M) {
  int row0 = blockIdx.x * 16;
  int wave = threadIdx.x >> 5;
  int lane = threadIdx.x & 31;
  int n0 = wave * 64;
  if (row0 >= M) return;

  f32x8 acc[4] = {};
  for (int k0 = 0; k0 < DD; k0 += 32) {
    bf16x16 a = load_a_f32(X, DD, row0, k0);
#pragma unroll
    for (int j = 0; j < 4; ++j) {
      bf16x16 b = load_b_bf16(Wt, DD, n0 + 16 * j, k0);
      acc[j] = wmma_bf16(a, b, acc[j]);
    }
  }
  int half = lane >> 4, col = lane & 15;
#pragma unroll
  for (int j = 0; j < 4; ++j) {
    int n = n0 + 16 * j + col;
    float bv = bias[n];
#pragma unroll
    for (int r = 0; r < 8; ++r) {
      int m = row0 + r + 8 * half;
      Y[(size_t)m * EE + n] = acc[j][r] + bv;
    }
  }
}

// one wave per row of 256: norms[row] = ||row||
__global__ void k_row_norms(const float* __restrict__ X, float* __restrict__ norms, int rows) {
  int w = (blockIdx.x * blockDim.x + threadIdx.x) >> 5;
  int lane = threadIdx.x & 31;
  if (w >= rows) return;
  const float* p = X + (size_t)w * EE;
  float ss = 0.f;
  for (int i = lane; i < EE; i += 32) { float x = p[i]; ss += x * x; }
  ss = wave_sum(ss);
  if (lane == 0) norms[w] = sqrtf(ss);
}

// one wave per token row: norms[row]=||tok|| ; sims[row]=dot(tok,cls)/(max(||tok||,eps)*max(||cls||,eps))
__global__ void k_tok_sim_norm(const float* __restrict__ proj, const float* __restrict__ clsproj,
                               const float* __restrict__ clsnorm, float* __restrict__ sims,
                               float* __restrict__ norms, int rows, int perB) {
  int w = (blockIdx.x * blockDim.x + threadIdx.x) >> 5;
  int lane = threadIdx.x & 31;
  if (w >= rows) return;
  int b = w / perB;
  const float* tp = proj + (size_t)w * EE;
  const float* cp = clsproj + (size_t)b * EE;
  float ss = 0.f, dp = 0.f;
  for (int i = lane; i < EE; i += 32) { float x = tp[i], c = cp[i]; ss += x * x; dp += x * c; }
  ss = wave_sum(ss);
  dp = wave_sum(dp);
  if (lane == 0) {
    float n = sqrtf(ss);
    norms[w] = n;
    sims[w] = dp / (fmaxf(n, 1e-12f) * fmaxf(clsnorm[b], 1e-12f));
  }
}

// one wave per row: iterative top-50 (descending; ties -> smaller index)
__global__ void k_topk(const float* __restrict__ sims, int* __restrict__ outIdx, int rowLen) {
  int row = (blockIdx.x * blockDim.x + threadIdx.x) >> 5;
  int lane = threadIdx.x & 31;
  if (row >= BN) return;
  const float* s = sims + (size_t)row * rowLen;
  float v[7];
#pragma unroll
  for (int j = 0; j < 7; ++j) {
    int id = lane + 32 * j;
    v[j] = (id < rowLen) ? s[id] : -1e30f;
  }
  for (int k = 0; k < KK; ++k) {
    float bv = -1e30f; int bi = 0x7fffffff;
#pragma unroll
    for (int j = 0; j < 7; ++j) {
      int id = lane + 32 * j;
      if (v[j] > bv || (v[j] == bv && id < bi)) { bv = v[j]; bi = id; }
    }
#pragma unroll
    for (int off = 16; off > 0; off >>= 1) {
      float ov = __shfl_xor(bv, off, 32);
      int   oi = __shfl_xor(bi, off, 32);
      if (ov > bv || (ov == bv && oi < bi)) { bv = ov; bi = oi; }
    }
    if (lane == 0) outIdx[row * KK + k] = bi;
    if ((bi & 31) == lane) v[bi >> 5] = -1e30f;
  }
}

// one wave per (b, k in 0..63): gather+normalize -> f32 out ; padded bf16 copy for WMMA
__global__ void k_gather(const float* __restrict__ proj, const float* __restrict__ norms,
                         const int* __restrict__ idx, float* __restrict__ outF,
                         __bf16* __restrict__ outB, int perB) {
  int w = (blockIdx.x * blockDim.x + threadIdx.x) >> 5;
  int lane = threadIdx.x & 31;
  if (w >= BN * KPAD) return;
  int b = w >> 6, k = w & 63;
  __bf16* ob = outB + ((size_t)b * KPAD + k) * EE;
  if (k < KK) {
    int src = idx[b * KK + k];
    size_t row = (size_t)b * perB + src;
    float inv = 1.0f / fmaxf(norms[row], 1e-12f);
    const float* p = proj + row * EE;
    float* o = outF + ((size_t)b * KK + k) * EE;
    for (int i = lane; i < EE; i += 32) {
      float x = p[i] * inv;
      o[i] = x;
      ob[i] = (__bf16)x;
    }
  } else {
    for (int i = lane; i < EE; i += 32) ob[i] = (__bf16)0.f;
  }
}

// Block = 512 threads = 16 waves covering a GBxGQ (b,q) tile.
// K staged in LDS in two 128-wide chunks: 8 tiles x 64x128 bf16 = 128 KB.
// Each wave computes the padded 64x64 S for its (b,q) pair, frags read from LDS,
// then masked max/mean reductions entirely in registers.
__global__ void k_cross_sim(const __bf16* __restrict__ Vt, const __bf16* __restrict__ Tt,
                            float* __restrict__ i2t, float* __restrict__ t2i) {
  extern __shared__ char smem_raw[];
  __bf16* As = (__bf16*)smem_raw;                       // GB tiles of [64][KC]
  __bf16* Bs = As + (size_t)GB * KPAD * KC;             // GQ tiles of [64][KC]

  const int nq = QN / GQ;
  int b0 = (blockIdx.x / nq) * GB;
  int q0 = (blockIdx.x % nq) * GQ;

  int tid  = threadIdx.x;
  int lane = tid & 31;
  int wave = tid >> 5;
  int tb = wave >> 2, tq = wave & 3;          // 16 waves -> 4x4 pairs
  int b = b0 + tb, q = q0 + tq;

  f32x8 acc[4][4] = {};

  // per-tile staging geometry: 64 rows x (KC bf16 = 256 B = 16 uint4) per chunk
  const int R16   = KC * 2 / 16;              // uint4 per row = 16
  const int TILE4 = KPAD * R16;               // uint4 per tile = 1024

  for (int kc = 0; kc < EE; kc += KC) {
    // ---- cooperative stage global -> LDS ----
    for (int i = tid; i < GB * TILE4; i += CS_THREADS) {
      int t = i / TILE4, rem = i % TILE4;
      int row = rem / R16, c = rem % R16;
      const uint4* src = (const uint4*)(Vt + (size_t)(b0 + t) * KPAD * EE + (size_t)row * EE + kc);
      ((uint4*)(As + (size_t)t * KPAD * KC + (size_t)row * KC))[c] = src[c];
    }
    for (int i = tid; i < GQ * TILE4; i += CS_THREADS) {
      int t = i / TILE4, rem = i % TILE4;
      int row = rem / R16, c = rem % R16;
      const uint4* src = (const uint4*)(Tt + (size_t)(q0 + t) * KPAD * EE + (size_t)row * EE + kc);
      ((uint4*)(Bs + (size_t)t * KPAD * KC + (size_t)row * KC))[c] = src[c];
    }
    __syncthreads();

    // ---- WMMA over this K chunk, fragments from LDS ----
    const __bf16* A  = As + (size_t)tb * KPAD * KC;
    const __bf16* Bm = Bs + (size_t)tq * KPAD * KC;
    for (int k0 = 0; k0 < KC; k0 += 32) {
      bf16x16 a[4], bb[4];
#pragma unroll
      for (int i = 0; i < 4; ++i) a[i] = load_a_bf16(A, KC, 16 * i, k0);
#pragma unroll
      for (int j = 0; j < 4; ++j) bb[j] = load_b_bf16(Bm, KC, 16 * j, k0);
#pragma unroll
      for (int i = 0; i < 4; ++i)
#pragma unroll
        for (int j = 0; j < 4; ++j) acc[i][j] = wmma_bf16(a[i], bb[j], acc[i][j]);
    }
    __syncthreads();
  }

  int half = lane >> 4, col = lane & 15;

  // image_to_text: max over t per v-row, mean over v
  float sum_i2t = 0.f;
#pragma unroll
  for (int i = 0; i < 4; ++i) {
#pragma unroll
    for (int r = 0; r < 8; ++r) {
      int vrow = 16 * i + 8 * half + r;
      float m = -1e30f;
#pragma unroll
      for (int j = 0; j < 4; ++j) {
        int t = 16 * j + col;
        float s = acc[i][j][r];
        if (t < KK) m = fmaxf(m, s);
      }
#pragma unroll
      for (int off = 1; off <= 8; off <<= 1) m = fmaxf(m, __shfl_xor(m, off, 32));
      if (vrow < KK) sum_i2t += m;
    }
  }
  sum_i2t += __shfl_xor(sum_i2t, 16, 32);  // combine row-halves

  // text_to_image: max over v per t-col, mean over t
  float sum_t2i = 0.f;
#pragma unroll
  for (int j = 0; j < 4; ++j) {
    int t = 16 * j + col;
    float m = -1e30f;
#pragma unroll
    for (int i = 0; i < 4; ++i)
#pragma unroll
      for (int r = 0; r < 8; ++r) {
        int vrow = 16 * i + 8 * half + r;
        if (vrow < KK) m = fmaxf(m, acc[i][j][r]);
      }
    m = fmaxf(m, __shfl_xor(m, 16, 32));  // combine row-halves (cols duplicated)
    if (t < KK) sum_t2i += m;
  }
#pragma unroll
  for (int off = 1; off <= 8; off <<= 1) sum_t2i += __shfl_xor(sum_t2i, off, 32);

  if (lane == 0) {
    i2t[(size_t)b * QN + q] = sum_i2t / (float)KK;
    t2i[(size_t)b * QN + q] = sum_t2i / (float)KK;
  }
}

// ---------------- host launcher ----------------
extern "C" void kernel_launch(void* const* d_in, const int* in_sizes, int n_in,
                              void* d_out, int out_size, void* d_ws, size_t ws_size,
                              hipStream_t stream) {
  const float* visual_cls    = (const float*)d_in[0];
  const float* visual_tokens = (const float*)d_in[1];
  const float* textual_cls   = (const float*)d_in[2];
  const float* textual_tokens= (const float*)d_in[3];
  const float* Wv     = (const float*)d_in[4];
  const float* bv     = (const float*)d_in[5];
  const float* Wt     = (const float*)d_in[6];
  const float* bt     = (const float*)d_in[7];
  const float* Wv_tok = (const float*)d_in[8];
  const float* bv_tok = (const float*)d_in[9];
  const float* Wt_tok = (const float*)d_in[10];
  const float* bt_tok = (const float*)d_in[11];

  float* out = (float*)d_out;
  float* o_vcls = out;                                   // [384,256]
  float* o_tcls = out + (size_t)BN * EE;                 // [384,256]
  float* o_vloc = out + 2 * (size_t)BN * EE;             // [384,50,256]
  float* o_tloc = o_vloc + (size_t)BN * KK * EE;         // [384,50,256]
  float* o_i2t  = o_tloc + (size_t)QN * KK * EE;         // [384,384]
  float* o_t2i  = o_i2t + (size_t)BN * QN;               // [384,384]

  // workspace carve-up (256B aligned)
  char* ws = (char*)d_ws;
  size_t cur = 0;
  auto carve = [&](size_t bytes) { char* p = ws + cur; cur += (bytes + 255) & ~(size_t)255; return p; };
  __bf16* wt_v    = (__bf16*)carve((size_t)EE * DD * 2);
  __bf16* wt_t    = (__bf16*)carve((size_t)EE * DD * 2);
  __bf16* wt_vtok = (__bf16*)carve((size_t)EE * DD * 2);
  __bf16* wt_ttok = (__bf16*)carve((size_t)EE * DD * 2);
  float* v_proj   = (float*)carve((size_t)BN * VN * EE * 4);
  float* t_proj   = (float*)carve((size_t)QN * TN * EE * 4);
  float* vcls_nrm = (float*)carve((size_t)BN * 4);
  float* tcls_nrm = (float*)carve((size_t)QN * 4);
  float* v_nrm    = (float*)carve((size_t)BN * VN * 4);
  float* t_nrm    = (float*)carve((size_t)QN * TN * 4);
  float* v_sim    = (float*)carve((size_t)BN * VN * 4);
  float* t_sim    = (float*)carve((size_t)QN * TN * 4);
  int*   iv       = (int*)carve((size_t)BN * KK * 4);
  int*   it       = (int*)carve((size_t)QN * KK * 4);
  __bf16* vloc_b  = (__bf16*)carve((size_t)BN * KPAD * EE * 2);
  __bf16* tloc_b  = (__bf16*)carve((size_t)QN * KPAD * EE * 2);
  (void)ws_size; (void)in_sizes; (void)n_in; (void)out_size;

  // 1) weights -> transposed bf16
  {
    int n = EE * DD, blk = 256, g = (n + blk - 1) / blk;
    k_transpose_w<<<g, blk, 0, stream>>>(Wv,     wt_v);
    k_transpose_w<<<g, blk, 0, stream>>>(Wt,     wt_t);
    k_transpose_w<<<g, blk, 0, stream>>>(Wv_tok, wt_vtok);
    k_transpose_w<<<g, blk, 0, stream>>>(Wt_tok, wt_ttok);
  }

  // 2) projections (WMMA)
  k_gemm_proj<<<BN / 16, 128, 0, stream>>>(visual_cls,  wt_v,    bv,     o_vcls, BN);
  k_gemm_proj<<<QN / 16, 128, 0, stream>>>(textual_cls, wt_t,    bt,     o_tcls, QN);
  k_gemm_proj<<<(BN * VN) / 16, 128, 0, stream>>>(visual_tokens,  wt_vtok, bv_tok, v_proj, BN * VN);
  k_gemm_proj<<<(QN * TN) / 16, 128, 0, stream>>>(textual_tokens, wt_ttok, bt_tok, t_proj, QN * TN);

  // 3) cls norms
  k_row_norms<<<(BN + 3) / 4, 128, 0, stream>>>(o_vcls, vcls_nrm, BN);
  k_row_norms<<<(QN + 3) / 4, 128, 0, stream>>>(o_tcls, tcls_nrm, QN);

  // 4) fused token norms + token->cls similarity
  k_tok_sim_norm<<<(BN * VN + 3) / 4, 128, 0, stream>>>(v_proj, o_vcls, vcls_nrm, v_sim, v_nrm, BN * VN, VN);
  k_tok_sim_norm<<<(QN * TN + 3) / 4, 128, 0, stream>>>(t_proj, o_tcls, tcls_nrm, t_sim, t_nrm, QN * TN, TN);

  // 5) top-k
  k_topk<<<(BN + 3) / 4, 128, 0, stream>>>(v_sim, iv, VN);
  k_topk<<<(QN + 3) / 4, 128, 0, stream>>>(t_sim, it, TN);

  // 6) gather + normalize (f32 outputs + zero-padded bf16 tiles)
  k_gather<<<(BN * KPAD + 3) / 4, 128, 0, stream>>>(v_proj, v_nrm, iv, o_vloc, vloc_b, VN);
  k_gather<<<(QN * KPAD + 3) / 4, 128, 0, stream>>>(t_proj, t_nrm, it, o_tloc, tloc_b, TN);

  // 7) all-pairs cross-modal sim + masked max/mean (WMMA, LDS-tiled 4x4 pair blocks)
  {
    int grid = (BN / GB) * (QN / GQ);                     // 9216 blocks
    size_t shmem = (size_t)(GB + GQ) * KPAD * KC * 2;     // 128 KB
    k_cross_sim<<<grid, CS_THREADS, shmem, stream>>>(vloc_b, tloc_b, o_i2t, o_t2i);
  }
}
